// Peft_ViT_14645838479987
// MI455X (gfx1250) — compile-verified, matrix operand order checked by reading, not run on hardware
//
#include <hip/hip_runtime.h>
#include <math.h>

typedef __bf16 bf16;
typedef __attribute__((ext_vector_type(16))) __bf16 v16bf;
typedef __attribute__((ext_vector_type(8)))  __bf16 v8bf;
typedef __attribute__((ext_vector_type(8)))  float  v8f;

#define DEV __device__ __forceinline__

DEV v8f zero_v8f() {
  v8f r;
#pragma unroll
  for (int i = 0; i < 8; ++i) r[i] = 0.0f;
  return r;
}

DEV v16bf zero_v16bf() {
  v16bf r;
#pragma unroll
  for (int i = 0; i < 16; ++i) r[i] = (bf16)0.0f;
  return r;
}

// LDS byte offset (wave-relative) from a generic pointer to a __shared__ object:
// generic LDS addresses carry the LDS offset in the low 32 bits.
DEV uint32_t lds_off(const void* p) { return (uint32_t)(uintptr_t)p; }

// Async 16B global -> LDS copy (CDNA5 VGLOBAL op 98, tracked by ASYNCcnt).
DEV void async_ld_b128(uint32_t ldsByteOff, const void* gaddr) {
  asm volatile("global_load_async_to_lds_b128 %0, %1, off"
               :
               : "v"(ldsByteOff), "v"(gaddr)
               : "memory");
}

DEV void wait_asynccnt0() { asm volatile("s_wait_asynccnt 0x0" ::: "memory"); }

// Load one 16x32 bf16 WMMA fragment (A-layout) / 32x16 (B-layout, symmetric).
// p must already point at: base + (rowBase + (lane&15))*stride + kBase + (lane>>4)*8
// Elements e=0..7  <- p[0..7]    (K = hi*8 + e)
// Elements e=8..15 <- p[16..23]  (K = 16 + hi*8 + e-8)
DEV v16bf load_frag(const bf16* p) {
  v8bf lo = *(const v8bf*)p;
  v8bf hh = *(const v8bf*)(p + 16);
  v16bf r;
#pragma unroll
  for (int i = 0; i < 8; ++i) { r[i] = lo[i]; r[i + 8] = hh[i]; }
  return r;
}

DEV v8f wmma_bf16(v16bf a, v16bf b, v8f c) {
  return __builtin_amdgcn_wmma_f32_16x16x32_bf16(false, a, false, b, (short)0, c,
                                                 false, false);
}

// ---------------------------------------------------------------------------
// Generic GEMM: out[M,N] = A[M,K](bf16) * W[N,K]^T(bf16)  (+bias, epilogue)
// Double-buffered LDS fed by async global->LDS DMA; one barrier per k-step.
// mode 0: f32 store (acc+bias)
// mode 1: f32 residual add (out += acc+bias)
// mode 2: bf16 store (acc+bias)
// mode 3: bf16 store gelu(acc+bias)   (exact gelu, erf)
// ---------------------------------------------------------------------------
__global__ __launch_bounds__(256) void k_gemm(const bf16* __restrict__ A,
                                              const bf16* __restrict__ W,
                                              const float* __restrict__ bias,
                                              float* __restrict__ outF,
                                              bf16* __restrict__ outH,
                                              int M, int N, int K, int mode) {
  __shared__ __align__(16) bf16 As[2][128 * 40];
  __shared__ __align__(16) bf16 Ws[2][128 * 40];

  const int tid  = threadIdx.x;
  const int lane = tid & 31, wave = tid >> 5;
  const int wm = wave & 3, wn = wave >> 2;   // 4 x 2 wave grid
  const int ln = lane & 15, hi = lane >> 4;
  const int rowBase = blockIdx.x * 128;
  const int colBase = blockIdx.y * 128;

  v8f acc[2][4];
#pragma unroll
  for (int mt = 0; mt < 2; ++mt)
#pragma unroll
    for (int nt = 0; nt < 4; ++nt) acc[mt][nt] = zero_v8f();

  // staging: 128 rows x 32 bf16 = 512 chunks of 16B; 2 A-chunks + 2 W-chunks /thread
  const int r0 = tid >> 2,         q0 = tid & 3;
  const int r1 = (tid + 256) >> 2, q1 = (tid + 256) & 3;

  auto stage = [&](int buf, int kk) {
    int gr0 = rowBase + r0, gr1 = rowBase + r1;
    if (gr0 < M)
      async_ld_b128(lds_off(&As[buf][r0 * 40 + q0 * 8]),
                    A + (size_t)gr0 * K + kk + q0 * 8);
    if (gr1 < M)
      async_ld_b128(lds_off(&As[buf][r1 * 40 + q1 * 8]),
                    A + (size_t)gr1 * K + kk + q1 * 8);
    async_ld_b128(lds_off(&Ws[buf][r0 * 40 + q0 * 8]),
                  W + (size_t)(colBase + r0) * K + kk + q0 * 8);
    async_ld_b128(lds_off(&Ws[buf][r1 * 40 + q1 * 8]),
                  W + (size_t)(colBase + r1) * K + kk + q1 * 8);
  };

  stage(0, 0);
  wait_asynccnt0();
  __syncthreads();

  const int nk = K >> 5;
  for (int ks = 0; ks < nk; ++ks) {
    const int cur = ks & 1;
    if (ks + 1 < nk) stage(cur ^ 1, (ks + 1) << 5);  // DMA overlaps WMMA below

    v16bf af[2], wf[4];
#pragma unroll
    for (int mt = 0; mt < 2; ++mt)
      af[mt] = load_frag(&As[cur][(wm * 32 + mt * 16 + ln) * 40 + hi * 8]);
#pragma unroll
    for (int nt = 0; nt < 4; ++nt)
      wf[nt] = load_frag(&Ws[cur][(wn * 64 + nt * 16 + ln) * 40 + hi * 8]);
#pragma unroll
    for (int mt = 0; mt < 2; ++mt)
#pragma unroll
      for (int nt = 0; nt < 4; ++nt)
        acc[mt][nt] = wmma_bf16(af[mt], wf[nt], acc[mt][nt]);

    wait_asynccnt0();   // next buffer's DMA complete (our wave)
    __syncthreads();    // ... and everyone else's; cur buffer free for reuse
  }

#pragma unroll
  for (int mt = 0; mt < 2; ++mt)
#pragma unroll
    for (int nt = 0; nt < 4; ++nt)
#pragma unroll
      for (int v = 0; v < 8; ++v) {
        int gr = rowBase + wm * 32 + mt * 16 + v + 8 * hi;
        int gc = colBase + wn * 64 + nt * 16 + ln;
        if (gr < M) {
          float xv = acc[mt][nt][v] + bias[gc];
          size_t idx = (size_t)gr * N + gc;
          if (mode == 0)      outF[idx] = xv;
          else if (mode == 1) outF[idx] += xv;
          else if (mode == 2) outH[idx] = (bf16)xv;
          else                outH[idx] = (bf16)(0.5f * xv * (1.0f + erff(xv * 0.70710678f)));
        }
      }
}

// ---------------------------------------------------------------------------
// Attention: one block per (b, head). S=213 padded to 224 (14 tiles).
// ---------------------------------------------------------------------------
__global__ __launch_bounds__(256) void k_attention(const bf16* __restrict__ qkv,
                                                   const float* __restrict__ coeff,
                                                   bf16* __restrict__ o) {
  __shared__ __align__(16) bf16 Vt[64 * 232];       // V transposed [dim][key]
  __shared__ __align__(16) bf16 slab[8][16 * 40];   // per-wave P relayout slab

  const int bh = blockIdx.x;
  const int b = bh / 12, hd = bh % 12;
  const bf16* qb = qkv + (size_t)b * 213 * 2304;

  for (int i = threadIdx.x; i < 64 * 224; i += 256) {
    int d = i / 224, j = i % 224;
    Vt[d * 232 + j] = (j < 213) ? qb[(size_t)j * 2304 + 1536 + hd * 64 + d]
                                : (bf16)0.0f;
  }
  __syncthreads();

  const int lane = threadIdx.x & 31, wave = threadIdx.x >> 5;
  const int ln = lane & 15, hi = lane >> 4;

  for (int qt = wave; qt < 14; qt += 8) {
    const int qs = qt * 16 + ln;
    v16bf qf[2];
#pragma unroll
    for (int kc = 0; kc < 2; ++kc)
      qf[kc] = (qs < 213)
                   ? load_frag(qb + (size_t)qs * 2304 + hd * 64 + kc * 32 + hi * 8)
                   : zero_v16bf();

    // scores: Q[16x64] x K^T -> 14 tiles of 16x16
    v8f st[14];
#pragma unroll
    for (int kt = 0; kt < 14; ++kt) {
      v8f a = zero_v8f();
      int ks = kt * 16 + ln;
#pragma unroll
      for (int kc = 0; kc < 2; ++kc) {
        v16bf kf = (ks < 213)
                       ? load_frag(qb + (size_t)ks * 2304 + 768 + hd * 64 + kc * 32 + hi * 8)
                       : zero_v16bf();
        a = wmma_bf16(qf[kc], kf, a);
      }
      st[kt] = a;
    }

    // scale * coeff, mask padded keys
#pragma unroll
    for (int kt = 0; kt < 14; ++kt) {
      int key = kt * 16 + ln;
      float cf = (key < 197) ? 1.0f : ((key < 213) ? coeff[key - 197] : 0.0f);
      bool msk = key >= 213;
#pragma unroll
      for (int v = 0; v < 8; ++v) {
        float xv = st[kt][v] * 0.125f * cf;
        st[kt][v] = msk ? -1e30f : xv;
      }
    }

    // softmax: row (v, half-wave) groups of 16 lanes
#pragma unroll
    for (int v = 0; v < 8; ++v) {
      float mx = -1e30f;
#pragma unroll
      for (int kt = 0; kt < 14; ++kt) mx = fmaxf(mx, st[kt][v]);
#pragma unroll
      for (int m = 1; m < 16; m <<= 1) mx = fmaxf(mx, __shfl_xor(mx, m, 32));
      float sum = 0.f;
#pragma unroll
      for (int kt = 0; kt < 14; ++kt) {
        float e = __expf(st[kt][v] - mx);
        st[kt][v] = e;
        sum += e;
      }
#pragma unroll
      for (int m = 1; m < 16; m <<= 1) sum += __shfl_xor(sum, m, 32);
      float inv = 1.0f / sum;
#pragma unroll
      for (int kt = 0; kt < 14; ++kt) st[kt][v] *= inv;
    }

    // O = P x V (drain 32-key chunks through wave-private slab)
    v8f oa[4];
#pragma unroll
    for (int nt = 0; nt < 4; ++nt) oa[nt] = zero_v8f();
    bf16* sl = slab[wave];
#pragma unroll
    for (int kc = 0; kc < 7; ++kc) {
#pragma unroll
      for (int v = 0; v < 8; ++v) {
        int ql = v + 8 * hi;                 // C-layout row -> slab row
        sl[ql * 40 + ln]      = (bf16)st[2 * kc][v];
        sl[ql * 40 + 16 + ln] = (bf16)st[2 * kc + 1][v];
      }
      v16bf pf = load_frag(sl + ln * 40 + hi * 8);
#pragma unroll
      for (int nt = 0; nt < 4; ++nt) {
        v16bf vf = load_frag(Vt + (size_t)(nt * 16 + ln) * 232 + kc * 32 + hi * 8);
        oa[nt] = wmma_bf16(pf, vf, oa[nt]);
      }
    }

#pragma unroll
    for (int nt = 0; nt < 4; ++nt)
#pragma unroll
      for (int v = 0; v < 8; ++v) {
        int s = qt * 16 + v + 8 * hi;
        if (s < 213)
          o[((size_t)b * 213 + s) * 768 + hd * 64 + nt * 16 + ln] = (bf16)oa[nt][v];
      }
  }
}

// ---------------------------------------------------------------------------
// LayerNorm (one wave per 768-wide row) -> bf16
// ---------------------------------------------------------------------------
__global__ __launch_bounds__(256) void k_ln_bf16(const float* __restrict__ x,
                                                 const float* __restrict__ g,
                                                 const float* __restrict__ b,
                                                 bf16* __restrict__ out, int rows) {
  int row = blockIdx.x * 8 + (threadIdx.x >> 5);
  int lane = threadIdx.x & 31;
  if (row >= rows) return;
  const float* xr = x + (size_t)row * 768;
  float v[24];
  float s = 0.f;
#pragma unroll
  for (int i = 0; i < 24; ++i) { v[i] = xr[lane + i * 32]; s += v[i]; }
#pragma unroll
  for (int m = 16; m >= 1; m >>= 1) s += __shfl_xor(s, m, 32);
  float mean = s * (1.0f / 768.0f);
  float q = 0.f;
#pragma unroll
  for (int i = 0; i < 24; ++i) { float d = v[i] - mean; q += d * d; }
#pragma unroll
  for (int m = 16; m >= 1; m >>= 1) q += __shfl_xor(q, m, 32);
  float rstd = rsqrtf(q * (1.0f / 768.0f) + 1e-6f);
  bf16* orow = out + (size_t)row * 768;
#pragma unroll
  for (int i = 0; i < 24; ++i) {
    int c = lane + i * 32;
    orow[c] = (bf16)((v[i] - mean) * rstd * g[c] + b[c]);
  }
}

// cls/patch + pos, ln_pre, write f32 residual stream (rows 0..196 of each b)
__global__ __launch_bounds__(256) void k_assemble(const float* __restrict__ patchOut,
                                                  const float* __restrict__ cls,
                                                  const float* __restrict__ pos,
                                                  const float* __restrict__ g,
                                                  const float* __restrict__ b,
                                                  float* __restrict__ h) {
  int row = blockIdx.x * 8 + (threadIdx.x >> 5);  // 0..6303 = b*197+s
  int lane = threadIdx.x & 31;
  int bb = row / 197, s = row % 197;
  size_t pidx = (s == 0) ? 0 : ((size_t)bb * 196 + (s - 1));
  const float* pr = patchOut + pidx * 768;
  float v[24];
  float sum = 0.f;
#pragma unroll
  for (int i = 0; i < 24; ++i) {
    int c = lane + i * 32;
    float base = (s == 0) ? cls[c] : pr[c];
    v[i] = base + pos[(size_t)s * 768 + c];
    sum += v[i];
  }
#pragma unroll
  for (int m = 16; m >= 1; m >>= 1) sum += __shfl_xor(sum, m, 32);
  float mean = sum * (1.0f / 768.0f);
  float q = 0.f;
#pragma unroll
  for (int i = 0; i < 24; ++i) { float d = v[i] - mean; q += d * d; }
#pragma unroll
  for (int m = 16; m >= 1; m >>= 1) q += __shfl_xor(q, m, 32);
  float rstd = rsqrtf(q * (1.0f / 768.0f) + 1e-6f);
  float* hr = h + ((size_t)bb * 213 + s) * 768;
#pragma unroll
  for (int i = 0; i < 24; ++i) {
    int c = lane + i * 32;
    hr[c] = (v[i] - mean) * rstd * g[c] + b[c];
  }
}

__global__ __launch_bounds__(256) void k_ln_post(const float* __restrict__ h,
                                                 const float* __restrict__ g,
                                                 const float* __restrict__ b,
                                                 float* __restrict__ out) {
  int row = blockIdx.x * 8 + (threadIdx.x >> 5);  // batch index
  int lane = threadIdx.x & 31;
  if (row >= 32) return;
  const float* xr = h + (size_t)row * 213 * 768;  // cls token row
  float v[24];
  float s = 0.f;
#pragma unroll
  for (int i = 0; i < 24; ++i) { v[i] = xr[lane + i * 32]; s += v[i]; }
#pragma unroll
  for (int m = 16; m >= 1; m >>= 1) s += __shfl_xor(s, m, 32);
  float mean = s * (1.0f / 768.0f);
  float q = 0.f;
#pragma unroll
  for (int i = 0; i < 24; ++i) { float d = v[i] - mean; q += d * d; }
#pragma unroll
  for (int m = 16; m >= 1; m >>= 1) q += __shfl_xor(q, m, 32);
  float rstd = rsqrtf(q * (1.0f / 768.0f) + 1e-6f);
#pragma unroll
  for (int i = 0; i < 24; ++i) {
    int c = lane + i * 32;
    out[(size_t)row * 768 + c] = (v[i] - mean) * rstd * g[c] + b[c];
  }
}

__global__ void k_cvt(const float* __restrict__ a, bf16* __restrict__ o, int n) {
  int i = blockIdx.x * 256 + threadIdx.x;
  if (i < n) o[i] = (bf16)a[i];
}

__global__ void k_im2col(const float* __restrict__ x, bf16* __restrict__ o) {
  int i = blockIdx.x * 256 + threadIdx.x;
  if (i >= 6272 * 768) return;
  int r = i / 768, c = i % 768;
  int b = r / 196, p = r % 196;
  int py = p / 14, px = p % 14;
  int ch = c >> 8, rem = c & 255;
  int ii = rem >> 4, jj = rem & 15;
  o[i] = (bf16)x[(((size_t)b * 3 + ch) * 224 + py * 16 + ii) * 224 + px * 16 + jj];
}

__global__ void k_set_prompts(const float* __restrict__ pr, float* __restrict__ h) {
  int i = blockIdx.x * 256 + threadIdx.x;
  if (i >= 32 * 16 * 768) return;
  int b = i / (16 * 768);
  int rem = i % (16 * 768);
  h[((size_t)b * 213 + 197) * 768 + rem] = pr[rem];
}

// ---------------------------------------------------------------------------
extern "C" void kernel_launch(void* const* d_in, const int* in_sizes, int n_in,
                              void* d_out, int out_size, void* d_ws, size_t ws_size,
                              hipStream_t stream) {
  (void)in_sizes; (void)n_in; (void)out_size; (void)ws_size;
  const float* x        = (const float*)d_in[0];
  const float* coeff    = (const float*)d_in[1];   // [12,1,16]
  const float* patch_w  = (const float*)d_in[2];
  const float* patch_b  = (const float*)d_in[3];
  const float* cls      = (const float*)d_in[4];
  const float* pos      = (const float*)d_in[5];
  const float* lnpre_g  = (const float*)d_in[6];
  const float* lnpre_b  = (const float*)d_in[7];
  const float* prompts  = (const float*)d_in[8];   // [12,16,768]
  const float* ln1_g    = (const float*)d_in[9];
  const float* ln1_b    = (const float*)d_in[10];
  const float* qkv_w    = (const float*)d_in[11];  // [12,2304,768]
  const float* qkv_b    = (const float*)d_in[12];
  const float* proj_w   = (const float*)d_in[13];
  const float* proj_b   = (const float*)d_in[14];
  const float* ln2_g    = (const float*)d_in[15];
  const float* ln2_b    = (const float*)d_in[16];
  const float* fc1_w    = (const float*)d_in[17];  // [12,3072,768]
  const float* fc1_b    = (const float*)d_in[18];
  const float* fc2_w    = (const float*)d_in[19];  // [12,768,3072]
  const float* fc2_b    = (const float*)d_in[20];
  const float* lnpost_g = (const float*)d_in[21];
  const float* lnpost_b = (const float*)d_in[22];

  size_t off = 0;
  auto take = [&](size_t bytes) -> void* {
    void* p = (char*)d_ws + off;
    off += (bytes + 255) & ~(size_t)255;
    return p;
  };
  float* h    = (float*)take((size_t)6816 * 768 * 4);
  bf16*  hh   = (bf16*) take((size_t)6816 * 768 * 2);
  bf16*  oB   = (bf16*) take((size_t)6816 * 768 * 2);
  bf16*  Wbuf = (bf16*) take((size_t)3072 * 768 * 2);
  bf16*  qkvB = (bf16*) take((size_t)6816 * 2304 * 2);
  bf16*  ffn  = (bf16*) take((size_t)6816 * 3072 * 2);
  bf16*  Apatch   = qkvB;        // alias: used only before the layer loop
  float* patchOut = (float*)ffn; // alias: used only before the layer loop

  dim3 T(256);
  auto cdiv = [](int a, int b) { return (a + b - 1) / b; };

  // patch embedding as GEMM
  k_cvt<<<cdiv(768 * 768, 256), T, 0, stream>>>(patch_w, Wbuf, 768 * 768);
  k_im2col<<<cdiv(6272 * 768, 256), T, 0, stream>>>(x, Apatch);
  k_gemm<<<dim3(49, 6), T, 0, stream>>>(Apatch, Wbuf, patch_b, patchOut, nullptr,
                                        6272, 768, 768, 0);
  k_assemble<<<788, T, 0, stream>>>(patchOut, cls, pos, lnpre_g, lnpre_b, h);

  for (int l = 0; l < 12; ++l) {
    k_set_prompts<<<cdiv(32 * 16 * 768, 256), T, 0, stream>>>(
        prompts + (size_t)l * 16 * 768, h);
    k_ln_bf16<<<852, T, 0, stream>>>(h, ln1_g + l * 768, ln1_b + l * 768, hh, 6816);

    k_cvt<<<cdiv(2304 * 768, 256), T, 0, stream>>>(qkv_w + (size_t)l * 2304 * 768,
                                                   Wbuf, 2304 * 768);
    k_gemm<<<dim3(54, 18), T, 0, stream>>>(hh, Wbuf, qkv_b + l * 2304, nullptr,
                                           qkvB, 6816, 2304, 768, 2);

    k_attention<<<384, T, 0, stream>>>(qkvB, coeff + l * 16, oB);

    k_cvt<<<cdiv(768 * 768, 256), T, 0, stream>>>(proj_w + (size_t)l * 768 * 768,
                                                  Wbuf, 768 * 768);
    k_gemm<<<dim3(54, 6), T, 0, stream>>>(oB, Wbuf, proj_b + l * 768, h, nullptr,
                                          6816, 768, 768, 1);

    k_ln_bf16<<<852, T, 0, stream>>>(h, ln2_g + l * 768, ln2_b + l * 768, hh, 6816);

    k_cvt<<<cdiv(3072 * 768, 256), T, 0, stream>>>(fc1_w + (size_t)l * 3072 * 768,
                                                   Wbuf, 3072 * 768);
    k_gemm<<<dim3(54, 24), T, 0, stream>>>(hh, Wbuf, fc1_b + l * 3072, nullptr,
                                           ffn, 6816, 3072, 768, 3);

    k_cvt<<<cdiv(768 * 3072, 256), T, 0, stream>>>(fc2_w + (size_t)l * 768 * 3072,
                                                   Wbuf, 768 * 3072);
    k_gemm<<<dim3(54, 6), T, 0, stream>>>(ffn, Wbuf, fc2_b + l * 768, h, nullptr,
                                          6816, 768, 3072, 1);
  }

  k_ln_post<<<4, T, 0, stream>>>(h, lnpost_g, lnpost_b, (float*)d_out);
}
